// Fairy2Linear_87471303951009
// MI455X (gfx1250) — matrix operand order repaired and checked by reading, not compiled, not run on hardware
//
#include <hip/hip_runtime.h>
#include <stdint.h>

typedef _Float16 half_t;
typedef __attribute__((ext_vector_type(16))) _Float16 v16h;
typedef __attribute__((ext_vector_type(8)))  _Float16 v8h;
typedef __attribute__((ext_vector_type(8)))  float    v8f;
typedef int v4i __attribute__((vector_size(16)));

#define AS1 __attribute__((address_space(1)))
#define AS3 __attribute__((address_space(3)))

#define CIN   512
#define COUT  512
#define KDIM  1024   // 2*CIN
#define NDIM  1024   // 2*COUT
#define QMAXF 127.0f
#define QMINF (-128.0f)
#define EPSF  1e-5f

#define TILE_M 128
#define TILE_N 128
#define TILE_K 32
#define LDSA_STRIDE 40   // halves; 80B row stride (16B-aligned, bank-skewed)
#define LDSB_STRIDE 40

#if __has_builtin(__builtin_amdgcn_global_load_async_to_lds_b128)
#define USE_ASYNC 1
#else
#define USE_ASYNC 0
#endif

__device__ __forceinline__ void wait_async0() {
#if __has_builtin(__builtin_amdgcn_s_wait_asynccnt)
    __builtin_amdgcn_s_wait_asynccnt(0);
#else
    asm volatile("s_wait_asynccnt 0x0" ::: "memory");
#endif
}

__device__ __forceinline__ float sgnf(float x) {
    return (x > 0.f) ? 1.f : ((x < 0.f) ? -1.f : 0.f);
}

__device__ __forceinline__ half_t q8(float v, float s) {
    float q = rintf(v * s);                 // RNE, matches jnp.round
    q = fminf(fmaxf(q, QMINF), QMAXF);
    return (half_t)q;                       // exact: integer in [-128,127]
}

// quantize 16 fp32 -> integer-valued fp16, store 32B to LDS
__device__ __forceinline__ void store_q_tile(half_t* dst, float4 a0, float4 a1,
                                             float4 a2, float4 a3, float s) {
    v8h h0, h1;
    h0[0] = q8(a0.x, s); h0[1] = q8(a0.y, s); h0[2] = q8(a0.z, s); h0[3] = q8(a0.w, s);
    h0[4] = q8(a1.x, s); h0[5] = q8(a1.y, s); h0[6] = q8(a1.z, s); h0[7] = q8(a1.w, s);
    h1[0] = q8(a2.x, s); h1[1] = q8(a2.y, s); h1[2] = q8(a2.z, s); h1[3] = q8(a2.w, s);
    h1[4] = q8(a3.x, s); h1[5] = q8(a3.y, s); h1[6] = q8(a3.z, s); h1[7] = q8(a3.w, s);
    *(v8h*)dst       = h0;
    *(v8h*)(dst + 8) = h1;
}

// Reconstruct 2-stage residual phase quantization from the two global scales.
__device__ __forceinline__ void residual2(float re, float im, float s1, float s2,
                                          float& qre, float& qim) {
    bool  d1   = fabsf(re) >= fabsf(im);
    float q1re = d1 ? sgnf(re) * s1 : 0.f;
    float q1im = d1 ? 0.f : sgnf(im) * s1;
    float rre  = re - q1re;
    float rim  = im - q1im;
    bool  d2   = fabsf(rre) >= fabsf(rim);
    qre = q1re + (d2 ? sgnf(rre) * s2 : 0.f);
    qim = q1im + (d2 ? 0.f : sgnf(rim) * s2);
}

// ---------------- Kernel 1: per-token quant scale (127/absmax) ----------------
__global__ void k_rowscale(const float* __restrict__ x, float* __restrict__ qscale, int rows) {
    const int wave = threadIdx.x >> 5;
    const int lane = threadIdx.x & 31;
    const int row  = blockIdx.x * 8 + wave;
    if (row >= rows) return;
    const float4* p = (const float4*)(x + (size_t)row * KDIM);
    float m = 0.f;
#pragma unroll
    for (int j = 0; j < 8; ++j) {               // 1024 floats = 32 lanes * 8 float4
        float4 v = p[lane + j * 32];
        m = fmaxf(m, fmaxf(fmaxf(fabsf(v.x), fabsf(v.y)), fmaxf(fabsf(v.z), fabsf(v.w))));
    }
#pragma unroll
    for (int off = 16; off > 0; off >>= 1)
        m = fmaxf(m, __shfl_xor(m, off, 32));
    if (lane == 0) {
        m = fmaxf(m, EPSF);
        qscale[row] = QMAXF / m;
    }
}

// ---------------- Kernel 2: two-stage phase-quant scales (mean reductions) ----
__global__ void k_wscales(const float* __restrict__ Ure, const float* __restrict__ Uim,
                          const float* __restrict__ Wre, const float* __restrict__ Wim,
                          float* __restrict__ scales /*4 floats: s1U s2U s1W s2W*/) {
    const float* re = blockIdx.x ? Wre : Ure;
    const float* im = blockIdx.x ? Wim : Uim;
    __shared__ float red[256];
    __shared__ float s1sh;
    const int N = CIN * COUT;

    float acc = 0.f;
    for (int i = threadIdx.x; i < N; i += 256) {
        float ar = fabsf(re[i]), am = fabsf(im[i]);
        acc += (ar >= am) ? ar : am;
    }
    red[threadIdx.x] = acc;
    __syncthreads();
    for (int s = 128; s > 0; s >>= 1) {
        if (threadIdx.x < s) red[threadIdx.x] += red[threadIdx.x + s];
        __syncthreads();
    }
    if (threadIdx.x == 0) s1sh = red[0] * (1.f / (float)N);
    __syncthreads();
    const float s1 = s1sh;

    acc = 0.f;
    for (int i = threadIdx.x; i < N; i += 256) {
        float r = re[i], m = im[i];
        bool  dom = fabsf(r) >= fabsf(m);
        float rr = r - (dom ? sgnf(r) * s1 : 0.f);
        float rm = m - (dom ? 0.f : sgnf(m) * s1);
        float arr = fabsf(rr), arm = fabsf(rm);
        acc += (arr >= arm) ? arr : arm;
    }
    red[threadIdx.x] = acc;
    __syncthreads();
    for (int s = 128; s > 0; s >>= 1) {
        if (threadIdx.x < s) red[threadIdx.x] += red[threadIdx.x + s];
        __syncthreads();
    }
    if (threadIdx.x == 0) {
        scales[blockIdx.x * 2 + 0] = s1;
        scales[blockIdx.x * 2 + 1] = red[0] * (1.f / (float)N);
    }
}

// ---------------- Kernel 3: build fused fp16 weight matrix Wbig[n][k] ---------
__global__ void k_build_wbig(const float* __restrict__ Ure, const float* __restrict__ Uim,
                             const float* __restrict__ Wre, const float* __restrict__ Wim,
                             const float* __restrict__ scales, half_t* __restrict__ Wbig) {
    const int idx = blockIdx.x * 256 + threadIdx.x;  // 0..262143
    const int o = idx >> 9;
    const int i = idx & 511;
    const float s1u = scales[0], s2u = scales[1];
    const float s1w = scales[2], s2w = scales[3];

    float uqre, uqim, wqre, wqim;
    residual2(Ure[idx], Uim[idx], s1u, s2u, uqre, uqim);
    residual2(Wre[idx], Wim[idx], s1w, s2w, wqre, wqim);

    Wbig[(size_t)o * KDIM + i]                  = (half_t)(uqre + wqre);
    Wbig[(size_t)o * KDIM + CIN + i]            = (half_t)(wqim - uqim);
    Wbig[(size_t)(COUT + o) * KDIM + i]         = (half_t)(uqim + wqim);
    Wbig[(size_t)(COUT + o) * KDIM + CIN + i]   = (half_t)(uqre - wqre);
}

// ---------------- Kernel 4: WMMA GEMM: y = diag(dq) * (xq_int @ Wbig^T) -------
// Double-buffered LDS pipeline; B tile staged with async-to-LDS when available.
__launch_bounds__(256, 2)
__global__ void k_gemm(const float* __restrict__ x, const half_t* __restrict__ Wbig,
                       const float* __restrict__ qscale, float* __restrict__ y) {
    __shared__ __align__(16) half_t As[2][TILE_M * LDSA_STRIDE];
    __shared__ __align__(16) half_t Bs[2][TILE_N * LDSB_STRIDE];
    __shared__ float  qs[TILE_M];
    __shared__ float  dqs[TILE_M];

    const int m0   = blockIdx.y * TILE_M;
    const int n0   = blockIdx.x * TILE_N;
    const int t    = threadIdx.x;
    const int lane = t & 31;
    const int wave = t >> 5;
    const int wm   = wave & 3;   // M offset 32*wm
    const int wn   = wave >> 2;  // N offset 64*wn

    if (t < TILE_M) {
        float q = qscale[m0 + t];
        qs[t]  = q;
        dqs[t] = 1.0f / q;   // absmax/127
    }

    // tile-load coordinates: 2 threads per row, 16 halves each
    const int r  = t >> 1;
    const int kc = (t & 1) * 16;
    const float4* xrow = (const float4*)(x + (size_t)(m0 + r) * KDIM + kc); // +8 float4 per K step
    const half_t* wrow = Wbig + (size_t)(n0 + r) * KDIM + kc;
    half_t* lA0 = &As[0][r * LDSA_STRIDE + kc];
    half_t* lA1 = &As[1][r * LDSA_STRIDE + kc];
    half_t* lB0 = &Bs[0][r * LDSB_STRIDE + kc];
    half_t* lB1 = &Bs[1][r * LDSB_STRIDE + kc];

    // ---- prologue: stage tile 0 ----
#if USE_ASYNC
    __builtin_amdgcn_global_load_async_to_lds_b128((AS1 v4i*)(wrow),     (AS3 v4i*)lB0, 0, 0);
    __builtin_amdgcn_global_load_async_to_lds_b128((AS1 v4i*)(wrow + 8), (AS3 v4i*)(lB0 + 8), 0, 0);
#else
    v8h pb0 = ((const v8h*)wrow)[0];
    v8h pb1 = ((const v8h*)wrow)[1];
#endif
    float4 a0 = xrow[0], a1 = xrow[1], a2 = xrow[2], a3 = xrow[3];

    __syncthreads();                       // qs/dqs visible
    const float qrow = qs[r];

    store_q_tile(lA0, a0, a1, a2, a3, qrow);
#if USE_ASYNC
    wait_async0();
#else
    *(v8h*)lB0       = pb0;
    *(v8h*)(lB0 + 8) = pb1;
#endif
    __syncthreads();                       // tile 0 ready

    const v8f vzero = {0.f, 0.f, 0.f, 0.f, 0.f, 0.f, 0.f, 0.f};
    v8f acc[2][4];
#pragma unroll
    for (int i = 0; i < 2; ++i)
#pragma unroll
        for (int j = 0; j < 4; ++j) acc[i][j] = vzero;

    // fragment coordinates (documented 16-bit A/B layouts, wave32)
    const int rowA = lane & 15;
    const int kloA = (lane >> 4) * 8;    // A: K chunks {klo..klo+7, klo+16..klo+23}
    const int colB = lane & 15;
    const int kbB  = (lane >> 4) * 8;    // B: K chunks {kb..kb+7, kb+16..kb+23}

    const int KT = KDIM / TILE_K;
    for (int kt = 0; kt < KT; ++kt) {
        const int  cur  = kt & 1;
        const bool more = (kt + 1 < KT);
        half_t* lAn = (cur ? lA0 : lA1);
        half_t* lBn = (cur ? lB0 : lB1);

        // ---- stage kt+1: issue global traffic before the WMMA block ----
#if USE_ASYNC
        if (more) {
            const half_t* gp = wrow + (kt + 1) * TILE_K;
            __builtin_amdgcn_global_load_async_to_lds_b128((AS1 v4i*)(gp),     (AS3 v4i*)lBn, 0, 0);
            __builtin_amdgcn_global_load_async_to_lds_b128((AS1 v4i*)(gp + 8), (AS3 v4i*)(lBn + 8), 0, 0);
        }
#else
        v8h nb0, nb1;
        if (more) {
            const half_t* gp = wrow + (kt + 1) * TILE_K;
            nb0 = ((const v8h*)gp)[0];
            nb1 = ((const v8h*)gp)[1];
        }
#endif
        float4 na0, na1, na2, na3;
        if (more) {
            const float4* p = xrow + (kt + 1) * 8;
            na0 = p[0]; na1 = p[1]; na2 = p[2]; na3 = p[3];
        }
        if (kt + 2 < KT)                   // keep next-next x slab moving toward L2
            __builtin_prefetch(xrow + (kt + 2) * 8, 0, 1);

        // ---- fragments from buffer[cur] ----
        const half_t* Ac = As[cur];
        const half_t* Bc = Bs[cur];
        v16h Af[2];
#pragma unroll
        for (int fi = 0; fi < 2; ++fi) {
            const half_t* pa = Ac + (wm * 32 + fi * 16 + rowA) * LDSA_STRIDE + kloA;
            v8h lo = *(const v8h*)pa;
            v8h hi = *(const v8h*)(pa + 16);
            Af[fi] = __builtin_shufflevector(lo, hi, 0,1,2,3,4,5,6,7,8,9,10,11,12,13,14,15);
        }
        v16h Bf[4];
#pragma unroll
        for (int fj = 0; fj < 4; ++fj) {
            const half_t* pb = Bc + (wn * 64 + fj * 16 + colB) * LDSB_STRIDE + kbB;
            v8h lo = *(const v8h*)pb;
            v8h hi = *(const v8h*)(pb + 16);
            Bf[fj] = __builtin_shufflevector(lo, hi, 0,1,2,3,4,5,6,7,8,9,10,11,12,13,14,15);
        }

#pragma unroll
        for (int fi = 0; fi < 2; ++fi)
#pragma unroll
            for (int fj = 0; fj < 4; ++fj)
                acc[fi][fj] = __builtin_amdgcn_wmma_f32_16x16x32_f16(
                    false, Af[fi], false, Bf[fj], (short)0, acc[fi][fj], false, false);

        // ---- finish staging kt+1 ----
        if (more) {
            store_q_tile(lAn, na0, na1, na2, na3, qrow);
#if !USE_ASYNC
            *(v8h*)lBn       = nb0;
            *(v8h*)(lBn + 8) = nb1;
#endif
#if USE_ASYNC
            wait_async0();
#endif
            __syncthreads();               // tile kt+1 ready, tile kt fully consumed
        }
    }

    // ---- epilogue: dequantize rows and store ----
    const int rhi = (lane >> 4) * 8;
    const int col = lane & 15;
#pragma unroll
    for (int fi = 0; fi < 2; ++fi) {
#pragma unroll
        for (int fj = 0; fj < 4; ++fj) {
            v8f c = acc[fi][fj];
#pragma unroll
            for (int v = 0; v < 8; ++v) {
                int mloc = wm * 32 + fi * 16 + rhi + v;
                int n    = n0 + wn * 64 + fj * 16 + col;
                y[(size_t)(m0 + mloc) * NDIM + n] = c[v] * dqs[mloc];
            }
        }
    }
}

// ---------------- host-side launch ----------------
extern "C" void kernel_launch(void* const* d_in, const int* in_sizes, int n_in,
                              void* d_out, int out_size, void* d_ws, size_t ws_size,
                              hipStream_t stream) {
    (void)n_in; (void)out_size; (void)ws_size;
    const float* x   = (const float*)d_in[0];
    const float* Ure = (const float*)d_in[1];
    const float* Uim = (const float*)d_in[2];
    const float* Wre = (const float*)d_in[3];
    const float* Wim = (const float*)d_in[4];
    float* y = (float*)d_out;

    const int rows = in_sizes[0] / KDIM;   // 8*4096 = 32768

    // workspace layout: [0,16) scale scalars | [256, 256+rows*4) row qscales | Wbig fp16
    float* scales  = (float*)d_ws;
    float* qscale  = (float*)((char*)d_ws + 256);
    size_t wb_off  = 256 + (((size_t)rows * 4 + 255) / 256) * 256;
    half_t* Wbig   = (half_t*)((char*)d_ws + wb_off);

    k_rowscale<<<(rows + 7) / 8, 256, 0, stream>>>(x, qscale, rows);
    k_wscales<<<2, 256, 0, stream>>>(Ure, Uim, Wre, Wim, scales);
    k_build_wbig<<<(CIN * COUT) / 256, 256, 0, stream>>>(Ure, Uim, Wre, Wim, scales, Wbig);

    dim3 grid(NDIM / TILE_N, rows / TILE_M);   // (8, 256): N fastest for L2 reuse of A
    k_gemm<<<grid, 256, 0, stream>>>(x, Wbig, qscale, y);
}
